// SEGCN_Layer_45732811768486
// MI455X (gfx1250) — matrix operand order repaired and checked by reading, not compiled, not run on hardware
//
#include <hip/hip_runtime.h>

// ---------------------------------------------------------------------------
// SEGCN layer for MI455X (gfx1250, wave32).
// Dense MLPs use v_wmma_f32_16x16x32_bf16 (f32 accumulate); sparse Chebyshev
// propagation uses float4 gathers + global_atomic_add_f32 scatters.
// ---------------------------------------------------------------------------

#define Nn 20000
#define Ee 320000
#define Hh 128
#define Kk 10
#define NH (Nn * Hh)

typedef __attribute__((ext_vector_type(16))) __bf16          v16bf;
typedef __attribute__((ext_vector_type(8)))  float           v8f;
typedef __attribute__((ext_vector_type(16))) unsigned short  v16u;
typedef __attribute__((ext_vector_type(8)))  unsigned short  u16x8;

__device__ __forceinline__ unsigned short f2bfu(float x) {
  union { float f; unsigned u; } v; v.f = x;
  unsigned r = v.u + 0x7FFFu + ((v.u >> 16) & 1u);   // round-to-nearest-even
  return (unsigned short)(r >> 16);
}
__device__ __forceinline__ float lrelu(float x) { return x > 0.f ? x : 0.02f * x; }

// A fragment: 16xK tile, row-major u16 LDS, per ISA 16-bit A 16x32 layout.
// lane<16: elems0-7=K[kb..kb+7], elems8-15=K[kb+16..kb+23]; lane>=16 shifted +8.
__device__ __forceinline__ v16bf load_frag_a(const unsigned short* base, int ld,
                                             int row, int kb, int hf) {
  const u16x8 lo = *(const u16x8*)(base + row * ld + kb + hf * 8);
  const u16x8 hi = *(const u16x8*)(base + row * ld + kb + 16 + hf * 8);
  v16u u;
#pragma unroll
  for (int j = 0; j < 8; ++j) { u[j] = lo[j]; u[j + 8] = hi[j]; }
  return __builtin_bit_cast(v16bf, u);
}

// B fragment from col-major (transposed, padded) bf16 weights: column n holds
// K per lane group: lane<16 -> K[kb..kb+7],K[kb+16..kb+23]; lane>=16 -> +8.
__device__ __forceinline__ v16bf load_frag_b(const unsigned short* wt, int ld,
                                             int n, int kb, int hf) {
  const unsigned short* p = wt + (size_t)n * ld + kb + hf * 8;
  const u16x8 b0 = *(const u16x8*)p;
  const u16x8 b1 = *(const u16x8*)(p + 16);
  v16u u;
#pragma unroll
  for (int j = 0; j < 8; ++j) { u[j] = b0[j]; u[j + 8] = b1[j]; }
  return __builtin_bit_cast(v16bf, u);
}

// ------------------------- small utility kernels ---------------------------

__global__ void zero_kernel(float* p, int n) {
  for (int i = blockIdx.x * blockDim.x + threadIdx.x; i < n; i += gridDim.x * blockDim.x)
    p[i] = 0.f;
}

__global__ void copy4_kernel(const float4* __restrict__ in, float4* __restrict__ out, int n4) {
  for (int i = blockIdx.x * blockDim.x + threadIdx.x; i < n4; i += gridDim.x * blockDim.x)
    out[i] = in[i];
}

__global__ void scale_store_kernel(float* __restrict__ out, const float* __restrict__ in,
                                   const float* __restrict__ temp, float binom, int ti, int n) {
  float c = binom * fmaxf(temp[ti], 0.f);
  for (int i = blockIdx.x * blockDim.x + threadIdx.x; i < n; i += gridDim.x * blockDim.x)
    out[i] = c * in[i];
}

__global__ void axpy_kernel(float* __restrict__ out, const float* __restrict__ in,
                            const float* __restrict__ temp, float binom, int ti, int n) {
  float c = binom * fmaxf(temp[ti], 0.f);
  for (int i = blockIdx.x * blockDim.x + threadIdx.x; i < n; i += gridDim.x * blockDim.x)
    out[i] += c * in[i];
}

__global__ void trelu_kernel(const float* __restrict__ temp, float* t1, float* t2) {
  int i = threadIdx.x;
  if (i < Kk + 1) { float v = fmaxf(temp[i], 0.f); t1[i] = v; t2[i] = v; }
}

// transpose KxNc row-major f32 -> Nc x ldp col-major bf16 (padded)
__global__ void prep_wt_kernel(const float* __restrict__ w, unsigned short* __restrict__ wt,
                               int K, int Nc, int ldp) {
  int total = K * Nc;
  for (int idx = blockIdx.x * blockDim.x + threadIdx.x; idx < total; idx += gridDim.x * blockDim.x) {
    int n = idx / K, k = idx - n * K;
    wt[(size_t)n * ldp + k] = f2bfu(w[(size_t)k * Nc + n]);
  }
}

// ------------------------ edge Gaussian MLP (wlap) -------------------------
// one wave per edge; lane handles 4 hidden units of the 128-wide layer.
__global__ __launch_bounds__(256) void edge_wlap_kernel(
    const float* __restrict__ x, const int* __restrict__ src, const int* __restrict__ dst,
    const float* __restrict__ eW1, const float* __restrict__ eb1,
    const float* __restrict__ eW2, const float* __restrict__ eb2,
    float* __restrict__ wlap, float* __restrict__ deg, float* __restrict__ cnt) {
  int wave = threadIdx.x >> 5, lane = threadIdx.x & 31;
  int e = blockIdx.x * 8 + wave;
  if (e >= Ee) return;
  int s = src[e], d = dst[e];
  float x0 = x[s * 3 + 0] - x[d * 3 + 0];
  float x1 = x[s * 3 + 1] - x[d * 3 + 1];
  float x2 = x[s * 3 + 2] - x[d * 3 + 2];
  float d2 = x0 * x0 + x1 * x1 + x2 * x2;
  float g[10]; float inv = 1.f;
#pragma unroll
  for (int i = 0; i < 10; ++i) { g[i] = __expf(-d2 * inv); inv *= 0.01f; }
  float part = 0.f;
#pragma unroll
  for (int uu = 0; uu < 4; ++uu) {
    int u = lane + uu * 32;
    float hsum = eb1[u];
#pragma unroll
    for (int i = 0; i < 10; ++i) hsum += g[i] * eW1[i * Hh + u];
    part += lrelu(hsum) * eW2[u];
  }
#pragma unroll
  for (int off = 16; off; off >>= 1) part += __shfl_xor(part, off, 32);
  if (lane == 0) {
    float w = part + eb2[0];
    w = w > 0.f ? w : 0.f;                   // relu
    wlap[e] = w;
    atomicAdd(&deg[s], w);
    atomicAdd(&cnt[d], 1.0f);
  }
}

// ----------------------- edge r-MLP (WMMA, bf16) ---------------------------
// 16 edges/block; 8 waves; each wave owns two 16x16 output tiles (32 of 256 cols).
#define LDC 264
__global__ __launch_bounds__(256) void edge_r_kernel(
    const float* __restrict__ h, const float* __restrict__ x,
    const int* __restrict__ src, const int* __restrict__ dst,
    const float* __restrict__ wlap, const unsigned short* __restrict__ rw1t,
    const float* __restrict__ rW1, const float* __restrict__ rb1,
    const float* __restrict__ rW2, const float* __restrict__ rb2,
    float* __restrict__ agg) {
  __shared__ unsigned short cat[16 * LDC];
  __shared__ float wl_s[16];
  __shared__ float rpart[16];
  __shared__ int   src_s[16], dst_s[16];

  int t = threadIdx.x;
  int eb = blockIdx.x * 16;
  if (t < 16) {
    int e = eb + t; if (e >= Ee) e = Ee - 1;
    src_s[t] = src[e]; dst_s[t] = dst[e];
    wl_s[t] = wlap[e]; rpart[t] = 0.f;
  }
  __syncthreads();
  // stage cat[m][0:256] = [h[src], h[dst]] as bf16, 1024 float4 loads total
  for (int q = t; q < 1024; q += 256) {
    int m = q >> 6, col = (q & 63) * 4;
    int node = (col < Hh) ? src_s[m] : dst_s[m];
    int off = (col < Hh) ? col : (col - Hh);
    float4 v = *(const float4*)(h + (size_t)node * Hh + off);
    unsigned short* p = cat + m * LDC + col;
    p[0] = f2bfu(v.x); p[1] = f2bfu(v.y); p[2] = f2bfu(v.z); p[3] = f2bfu(v.w);
  }
  __syncthreads();

  int wave = t >> 5, lane = t & 31, hf = lane >> 4, lcol = lane & 15;
  int nb0 = wave * 32;
  v8f acc0 = {}, acc1 = {};
#pragma unroll
  for (int kb = 0; kb < 256; kb += 32) {
    v16bf a = load_frag_a(cat, LDC, lcol, kb, hf);
    v16bf b0 = load_frag_b(rw1t, 264, nb0 + lcol, kb, hf);
    acc0 = __builtin_amdgcn_wmma_f32_16x16x32_bf16(false, a, false, b0, (short)0, acc0, false, false);
    v16bf b1 = load_frag_b(rw1t, 264, nb0 + 16 + lcol, kb, hf);
    acc1 = __builtin_amdgcn_wmma_f32_16x16x32_bf16(false, a, false, b1, (short)0, acc1, false, false);
  }
  // epilogue: add fp32 rank-1 wlap term (K=256 column), bias, lrelu, dot rW2
#pragma unroll
  for (int tt = 0; tt < 2; ++tt) {
    v8f acc = tt ? acc1 : acc0;
    int n = nb0 + tt * 16 + lcol;
    float w256 = rW1[256 * 256 + n];
    float b1v = rb1[n], w2 = rW2[n];
#pragma unroll
    for (int j = 0; j < 8; ++j) {
      int m = j + hf * 8;
      float v = acc[j] + wl_s[m] * w256 + b1v;
      atomicAdd(&rpart[m], lrelu(v) * w2);
    }
  }
  __syncthreads();
  if (t < 16 && eb + t < Ee) {
    float r = rpart[t] + rb2[0];
    int s = src_s[t], d = dst_s[t];
    atomicAdd(&agg[d * 3 + 0], r * (x[s * 3 + 0] - x[d * 3 + 0]));
    atomicAdd(&agg[d * 3 + 1], r * (x[s * 3 + 1] - x[d * 3 + 1]));
    atomicAdd(&agg[d * 3 + 2], r * (x[s * 3 + 2] - x[d * 3 + 2]));
  }
}

// ----------------------- node MLP h0 (WMMA, bf16) --------------------------
#define LDH 136
__global__ __launch_bounds__(256) void h0_kernel(
    const float* __restrict__ h, const unsigned short* __restrict__ fw1t,
    const float* __restrict__ fb1, const unsigned short* __restrict__ fw2t,
    const float* __restrict__ fb2, float* __restrict__ h0) {
  __shared__ unsigned short hA[16 * LDH];
  __shared__ unsigned short mid[16 * LDH];
  int t = threadIdx.x;
  int nb = blockIdx.x * 16;
  for (int q = t; q < 512; q += 256) {
    int m = q >> 5, col = (q & 31) * 4;
    int node = nb + m; if (node >= Nn) node = Nn - 1;
    float4 v = *(const float4*)(h + (size_t)node * Hh + col);
    unsigned short* p = hA + m * LDH + col;
    p[0] = f2bfu(v.x); p[1] = f2bfu(v.y); p[2] = f2bfu(v.z); p[3] = f2bfu(v.w);
  }
  __syncthreads();
  int wave = t >> 5, lane = t & 31, hf = lane >> 4, lcol = lane & 15;
  int n = wave * 16 + lcol;
  v8f acc = {};
#pragma unroll
  for (int kb = 0; kb < Hh; kb += 32) {
    v16bf a = load_frag_a(hA, LDH, lcol, kb, hf);
    v16bf b = load_frag_b(fw1t, LDH, n, kb, hf);
    acc = __builtin_amdgcn_wmma_f32_16x16x32_bf16(false, a, false, b, (short)0, acc, false, false);
  }
  float bv = fb1[n];
#pragma unroll
  for (int j = 0; j < 8; ++j) {
    int m = j + hf * 8;
    mid[m * LDH + n] = f2bfu(lrelu(acc[j] + bv));
  }
  __syncthreads();
  v8f acc2 = {};
#pragma unroll
  for (int kb = 0; kb < Hh; kb += 32) {
    v16bf a = load_frag_a(mid, LDH, lcol, kb, hf);
    v16bf b = load_frag_b(fw2t, LDH, n, kb, hf);
    acc2 = __builtin_amdgcn_wmma_f32_16x16x32_bf16(false, a, false, b, (short)0, acc2, false, false);
  }
  float b2 = fb2[n];
#pragma unroll
  for (int j = 0; j < 8; ++j) {
    int m = j + hf * 8, node = nb + m;
    if (node < Nn) h0[(size_t)node * Hh + n] = acc2[j] + b2;
  }
}

// ------------------------- node / edge finalize ----------------------------
__global__ void node_finalize_kernel(const float* __restrict__ x, const float* __restrict__ agg,
                                     const float* __restrict__ cnt, const float* __restrict__ deg,
                                     float* __restrict__ xout, float* __restrict__ dinv) {
  int i = blockIdx.x * blockDim.x + threadIdx.x;
  if (i >= Nn) return;
  float c = fmaxf(cnt[i], 1.f);
  xout[i * 3 + 0] = x[i * 3 + 0] + agg[i * 3 + 0] / c;
  xout[i * 3 + 1] = x[i * 3 + 1] + agg[i * 3 + 1] / c;
  xout[i * 3 + 2] = x[i * 3 + 2] + agg[i * 3 + 2] / c;
  float dg = deg[i];
  dinv[i] = dg > 0.f ? rsqrtf(dg) : 0.f;
}

__global__ void norm_e_kernel(const int* __restrict__ src, const int* __restrict__ dst,
                              const float* __restrict__ wlap, const float* __restrict__ dinv,
                              float* __restrict__ ne) {
  int e = blockIdx.x * blockDim.x + threadIdx.x;
  if (e < Ee) ne[e] = dinv[src[e]] * wlap[e] * dinv[dst[e]];
}

// -------------------- sparse propagation zout += s*A*zin --------------------
// one wave per edge; float4 gather, 4 global f32 atomics per lane.
__global__ __launch_bounds__(256) void spmv_kernel(
    const int* __restrict__ src, const int* __restrict__ dst, const float* __restrict__ ne,
    const float* __restrict__ zin, float* __restrict__ zout, float sign) {
  int e = blockIdx.x * 8 + (threadIdx.x >> 5);
  if (e >= Ee) return;
  float s = sign * ne[e];
  if (s == 0.f) return;
  int lane = threadIdx.x & 31;
  int sn = src[e], dn = dst[e];
  float4 v = *(const float4*)(zin + (size_t)sn * Hh + lane * 4);
  float* o = zout + (size_t)dn * Hh + lane * 4;
  atomicAdd(o + 0, s * v.x);
  atomicAdd(o + 1, s * v.y);
  atomicAdd(o + 2, s * v.z);
  atomicAdd(o + 3, s * v.w);
}

// ---------------------------------------------------------------------------

static const float BINOM_H[11] = {1.f / 1024, 10.f / 1024, 45.f / 1024, 120.f / 1024,
                                  210.f / 1024, 252.f / 1024, 210.f / 1024, 120.f / 1024,
                                  45.f / 1024, 10.f / 1024, 1.f / 1024};

struct Ws {
  unsigned short *rw1t, *fw1t, *fw2t;
  float *wlap, *ne, *deg, *cnt, *agg, *dinv, *tmp, *zA, *zB;
};

static void run_graph(const float* x, const float* h, const int* ei,
                      float* out_x, float* out_h,
                      const float* eW1, const float* eb1, const float* eW2, const float* eb2,
                      const float* rW1, const float* rb1, const float* rW2, const float* rb2,
                      const float* fb1, const float* fb2, const float* temp,
                      const Ws& w, hipStream_t st) {
  const int* src = ei;
  const int* dst = ei + Ee;
  // deg/cnt/agg are contiguous (5N floats)
  zero_kernel<<<512, 256, 0, st>>>(w.deg, 5 * Nn);
  edge_wlap_kernel<<<Ee / 8, 256, 0, st>>>(x, src, dst, eW1, eb1, eW2, eb2, w.wlap, w.deg, w.cnt);
  edge_r_kernel<<<Ee / 16, 256, 0, st>>>(h, x, src, dst, w.wlap, w.rw1t, rW1, rb1, rW2, rb2, w.agg);
  node_finalize_kernel<<<(Nn + 255) / 256, 256, 0, st>>>(x, w.agg, w.cnt, w.deg, out_x, w.dinv);
  norm_e_kernel<<<(Ee + 255) / 256, 256, 0, st>>>(src, dst, w.wlap, w.dinv, w.ne);
  h0_kernel<<<Nn / 16, 256, 0, st>>>(h, w.fw1t, fb1, w.fw2t, fb2, w.tmp);  // tmp[0] = h0

  // forward: tmp[k+1] = tmp[k] + A*tmp[k]
  for (int k = 0; k < Kk; ++k) {
    const float* zk = w.tmp + (size_t)k * NH;
    float* zk1 = w.tmp + (size_t)(k + 1) * NH;
    copy4_kernel<<<2500, 256, 0, st>>>((const float4*)zk, (float4*)zk1, NH / 4);
    spmv_kernel<<<Ee / 8, 256, 0, st>>>(src, dst, w.ne, zk, zk1, 1.f);
  }
  // out = BINOM[0]*relu(temp[0])*tmp[K]
  scale_store_kernel<<<2500, 256, 0, st>>>(out_h, w.tmp + (size_t)Kk * NH, temp, BINOM_H[0], 0, NH);
  // backward terms
  float* zA = w.zA;
  float* zB = w.zB;
  for (int i = 0; i < Kk; ++i) {
    copy4_kernel<<<2500, 256, 0, st>>>((const float4*)(w.tmp + (size_t)(Kk - i - 1) * NH),
                                       (float4*)zA, NH / 4);
    for (int tt = 0; tt <= i; ++tt) {
      copy4_kernel<<<2500, 256, 0, st>>>((const float4*)zA, (float4*)zB, NH / 4);
      spmv_kernel<<<Ee / 8, 256, 0, st>>>(src, dst, w.ne, zA, zB, -1.f);
      float* t2 = zA; zA = zB; zB = t2;
    }
    axpy_kernel<<<2500, 256, 0, st>>>(out_h, zA, temp, BINOM_H[i + 1], i + 1, NH);
  }
}

extern "C" void kernel_launch(void* const* d_in, const int* in_sizes, int n_in,
                              void* d_out, int out_size, void* d_ws, size_t ws_size,
                              hipStream_t stream) {
  const float* x1 = (const float*)d_in[0];
  const float* h1 = (const float*)d_in[1];
  const float* x2 = (const float*)d_in[2];
  const float* h2 = (const float*)d_in[3];
  const float* eW1 = (const float*)d_in[4];
  const float* eb1 = (const float*)d_in[5];
  const float* eW2 = (const float*)d_in[6];
  const float* eb2 = (const float*)d_in[7];
  const float* rW1 = (const float*)d_in[8];
  const float* rb1 = (const float*)d_in[9];
  const float* rW2 = (const float*)d_in[10];
  const float* rb2 = (const float*)d_in[11];
  const float* fW1 = (const float*)d_in[12];
  const float* fb1 = (const float*)d_in[13];
  const float* fW2 = (const float*)d_in[14];
  const float* fb2 = (const float*)d_in[15];
  const float* temp = (const float*)d_in[16];
  const int* ei1 = (const int*)d_in[17];
  const int* ei2 = (const int*)d_in[18];

  // output layout: x1n | h1n | x2n | h2n | T1 | T2
  float* out = (float*)d_out;
  float* out_x1 = out;
  float* out_h1 = out + Nn * 3;
  float* out_x2 = out_h1 + NH;
  float* out_h2 = out_x2 + Nn * 3;
  float* out_t1 = out_h2 + NH;
  float* out_t2 = out_t1 + (Kk + 1);

  // workspace carve (~137 MB)
  char* p = (char*)d_ws;
  auto carve = [&](size_t bytes) {
    void* r = (void*)p;
    p += (bytes + 255) & ~(size_t)255;
    return r;
  };
  Ws w;
  w.rw1t = (unsigned short*)carve((size_t)256 * 264 * 2);
  w.fw1t = (unsigned short*)carve((size_t)128 * LDH * 2);
  w.fw2t = (unsigned short*)carve((size_t)128 * LDH * 2);
  w.wlap = (float*)carve((size_t)Ee * 4);
  w.ne   = (float*)carve((size_t)Ee * 4);
  w.deg  = (float*)carve((size_t)5 * Nn * 4);  // deg | cnt | agg contiguous
  w.cnt  = w.deg + Nn;
  w.agg  = w.deg + 2 * Nn;
  w.dinv = (float*)carve((size_t)Nn * 4);
  w.tmp  = (float*)carve((size_t)(Kk + 1) * NH * 4);
  w.zA   = (float*)carve((size_t)NH * 4);
  w.zB   = (float*)carve((size_t)NH * 4);
  (void)ws_size; (void)in_sizes; (void)n_in; (void)out_size;

  // weight pre-transposes to padded col-major bf16
  prep_wt_kernel<<<512, 256, 0, stream>>>(rW1, w.rw1t, 257, 256, 264);
  prep_wt_kernel<<<128, 256, 0, stream>>>(fW1, w.fw1t, 128, 128, LDH);
  prep_wt_kernel<<<128, 256, 0, stream>>>(fW2, w.fw2t, 128, 128, LDH);
  trelu_kernel<<<1, 32, 0, stream>>>(temp, out_t1, out_t2);

  run_graph(x1, h1, ei1, out_x1, out_h1, eW1, eb1, eW2, eb2, rW1, rb1, rW2, rb2,
            fb1, fb2, temp, w, stream);
  run_graph(x2, h2, ei2, out_x2, out_h2, eW1, eb1, eW2, eb2, rW1, rb1, rW2, rb2,
            fb1, fb2, temp, w, stream);
}
